// GATWithPool_59433757442089
// MI455X (gfx1250) — matrix operand order repaired and checked by reading, not compiled, not run on hardware
//
#include <hip/hip_runtime.h>
#include <hip/hip_bf16.h>
#include <cmath>

// ---------------------------------------------------------------------------
// GAT-with-pool for MI455X (gfx1250, wave32, WMMA + TDM).
// - All matmuls use v_wmma_f32_16x16x32_f16 (f16 operands, f32 accumulate).
// - B-side operands kept TRANSPOSED (channel-major) so every WMMA B fragment
//   is one aligned 32B contiguous load; A fragments are two aligned 16B loads.
// - Edge-matrix tiles staged to LDS by the Tensor Data Mover (double-buffered,
//   s_wait_tensorcnt), descriptor built per ISA 8.3/8.4.
// - edge_attr (d_out+80) doubles as the masked agg matrix: >0 iff mask.
// ---------------------------------------------------------------------------

typedef __attribute__((ext_vector_type(16))) _Float16     v16h;
typedef __attribute__((ext_vector_type(8)))  _Float16     v8h;
typedef __attribute__((ext_vector_type(8)))  float        v8f;
typedef __attribute__((ext_vector_type(4)))  unsigned int v4u;
typedef __attribute__((ext_vector_type(8)))  int          v8i;
typedef __attribute__((ext_vector_type(4)))  int          v4i;

#define N_NODES 2048
#define NUM_G   8

#if __has_builtin(__builtin_amdgcn_tensor_load_to_lds)
#define HAVE_TDM 1
#else
#define HAVE_TDM 0
#endif

// ---------------- 1x1 conv over 12 attention maps + threshold --------------
__global__ void k_agg(const float* __restrict__ attn, const float* __restrict__ w_agg,
                      const float* __restrict__ b_agg, float* __restrict__ edge) {
  int idx = blockIdx.x * blockDim.x + threadIdx.x;            // 0..N*N-1
  const int NN = N_NODES * N_NODES;
  float acc = b_agg[0];
#pragma unroll
  for (int k = 0; k < 12; ++k) acc += w_agg[k] * attn[k * NN + idx];
  edge[idx] = acc > 0.f ? acc : 0.f;   // edge_attr == where(mask, agg, 0)
}

// -------- per-dst column stats: mean of positive off-diagonal values -------
__global__ void k_colstats(const float* __restrict__ edge, float* __restrict__ mean_e) {
  int j = blockIdx.x * blockDim.x + threadIdx.x;
  if (j >= N_NODES) return;
  float sum = 0.f; int cnt = 0;
  for (int i = 0; i < N_NODES; ++i) {
    float v = edge[i * N_NODES + j];
    if (v > 0.f && i != j) { sum += v; ++cnt; }
  }
  mean_e[j] = cnt > 0 ? sum / (float)cnt : 0.f;
}

// ---------------- fp32 -> fp16 conversions ---------------------------------
__global__ void k_f32tof16(const float* __restrict__ src, _Float16* __restrict__ dst, int n) {
  int i = blockIdx.x * blockDim.x + threadIdx.x;
  if (i < n) dst[i] = (_Float16)src[i];
}
// transpose convert: src [K][N] row-major -> dst [N][K]
__global__ void k_f32tof16_T(const float* __restrict__ src, _Float16* __restrict__ dst,
                             int K, int N) {
  int t = blockIdx.x * blockDim.x + threadIdx.x;
  if (t >= K * N) return;
  int k = t / N, n = t - k * N;
  dst[n * K + k] = (_Float16)src[t];
}

// ---------------- per-head edge weight: we[h] = sum_c We[h,c]*att_e[h,c] ---
__global__ void k_we(const float* __restrict__ We, const float* __restrict__ att_e,
                     float* __restrict__ we, int H, int C) {
  int h = threadIdx.x;
  if (h < H) {
    float s = 0.f;
    for (int c = 0; c < C; ++c) s += We[h * C + c] * att_e[h * C + c];
    we[h] = s;
  }
}

// ---------------- f16 GEMM via WMMA: C[M,N] = A[M,K] @ BT[N,K]^T -----------
template <bool WF32, bool WF16T>
__global__ void k_gemm_f16(const _Float16* __restrict__ A, const _Float16* __restrict__ BT,
                           float* __restrict__ Cf32, _Float16* __restrict__ Cf16T,
                           int M, int K, int Nout) {
  int m0 = blockIdx.x * 16;
  int n0 = blockIdx.y * 16;
  int lane = threadIdx.x & 31;
  int half = lane >> 4;
  int mrow = m0 + (lane & 15);
  int ncol = n0 + (lane & 15);
  int ah = half * 16;             // A K-offset for this lane group (contiguous k = 16*half + s)
  int bh = half * 16;             // B K-offset for this lane group
  v8f acc = {};
  const _Float16* arow = A + mrow * K;
  const _Float16* brow = BT + ncol * K;
  for (int k0 = 0; k0 < K; k0 += 32) {
    v8h alo = *(const v8h*)(arow + k0 + ah);
    v8h ahi = *(const v8h*)(arow + k0 + 8 + ah);
    v16h a = __builtin_shufflevector(alo, ahi, 0, 1, 2, 3, 4, 5, 6, 7,
                                     8, 9, 10, 11, 12, 13, 14, 15);
    v16h b = *(const v16h*)(brow + k0 + bh);
    acc = __builtin_amdgcn_wmma_f32_16x16x32_f16(false, a, false, b, (short)0, acc, false, false);
  }
  int nn = n0 + (lane & 15);
  if (WF32) {
#pragma unroll
    for (int r = 0; r < 8; ++r)
      Cf32[(m0 + r + (half ? 8 : 0)) * Nout + nn] = acc[r];
  }
  if (WF16T) {
    v8h o;
#pragma unroll
    for (int r = 0; r < 8; ++r) o[r] = (_Float16)acc[r];
    *(v8h*)(Cf16T + nn * M + m0 + 8 * half) = o;
  }
}

// -------- per-(node,head) attention scalars a_s, a_d -----------------------
__global__ void k_attdot(const float* __restrict__ hmat, const float* __restrict__ att_src,
                         const float* __restrict__ att_dst, float* __restrict__ a_s,
                         float* __restrict__ a_d, int H, int C) {
  int t = blockIdx.x * blockDim.x + threadIdx.x;   // n*H + h
  int n = t / H, h = t - n * H;
  if (n >= N_NODES) return;
  const float* row = hmat + n * H * C + h * C;
  float s = 0.f, d = 0.f;
  for (int c = 0; c < C; ++c) { s += row[c] * att_src[h * C + c]; d += row[c] * att_dst[h * C + c]; }
  a_s[t] = s; a_d[t] = d;
}

// ---- TDM: DMA a 32x16 f32 tile edge[s0..s0+31][j0..j0+15] into LDS --------
// Descriptor per ISA 8.3/8.4: count=1, type=2, data_size=4B, 2D tile.
__device__ __forceinline__ void edge_tile_issue(const float* __restrict__ edge,
                                                int j0, int s0, float* dstLds,
                                                int wave, int lane) {
  if (wave == 0) {
#if HAVE_TDM
    unsigned long long ga = (unsigned long long)(uintptr_t)(edge + s0 * N_NODES + j0);
    unsigned ldsOff = (unsigned)(uintptr_t)dstLds;      // low 32 bits = LDS offset
    v4u g0 = { 1u,                                      // count=1, user mode
               ldsOff,                                  // lds_addr
               (unsigned)ga,                            // global_addr[31:0]
               ((unsigned)(ga >> 32) & 0x01FFFFFFu) | 0x80000000u };  // addr[56:32] | type=2
    v8i g1 = { 0x20000,                                 // data_size=2 (4 bytes)
               (int)(N_NODES << 16),                    // tensor_dim0[15:0] @ bits63:48
               (int)(N_NODES << 16),                    // dim0 hi=0 | tensor_dim1[15:0]
               16 << 16,                                // dim1 hi=0 | tile_dim0=16
               32,                                      // tile_dim1=32, tile_dim2=0
               N_NODES,                                 // tensor_dim0_stride[31:0]
               0, 0 };
    v4i gz4 = { 0, 0, 0, 0 };
    v8i gz8 = { 0, 0, 0, 0, 0, 0, 0, 0 };
    __builtin_amdgcn_tensor_load_to_lds(g0, g1, gz4, gz4, gz8, 0);
#else
    for (int r = lane; r < 32 * 16; r += 32)
      dstLds[r] = edge[(s0 + (r >> 4)) * N_NODES + j0 + (r & 15)];
#endif
  }
}
__device__ __forceinline__ void edge_tile_wait_le1() {
#if HAVE_TDM
  __builtin_amdgcn_s_wait_tensorcnt(1);
#endif
}
__device__ __forceinline__ void edge_tile_wait_le0() {
#if HAVE_TDM
  __builtin_amdgcn_s_wait_tensorcnt(0);
#endif
}

// ---------------- GAT attention + weighted aggregation ---------------------
// Block = H waves; wave w = head w for one 16-dst tile. Edge tiles arrive via
// TDM (double-buffered). Pass 1: online softmax per dst column. Pass 2: alpha
// tile -> LDS (wave-private), WMMA against transposed f16 messages.
template <int H, int C, bool OUTF32, bool OUTF16>
__global__ void k_gat_attn(const float* __restrict__ edge,     // [N][N] masked agg
                           const float* __restrict__ mean_e,   // [N]
                           const float* __restrict__ a_s,      // [N][H]
                           const float* __restrict__ a_d,      // [N][H]
                           const float* __restrict__ we,       // [H]
                           const _Float16* __restrict__ hmsgT, // [H*C][N] messages^T
                           const float* __restrict__ bias,     // [H*C]
                           float* __restrict__ outf32,         // [N][H*C] (elu)
                           _Float16* __restrict__ outf16) {    // [N][H*C] (elu)
  constexpr int HC  = H * C;
  constexpr int NCH = N_NODES / 32;
  __shared__ _Float16 alphaT[H][16][32];   // [dst][src], wave-private per head
  __shared__ float    evt[2][32][16];      // TDM-staged edge tiles
  const int j0   = blockIdx.x * 16;
  const int wave = threadIdx.x >> 5;       // head
  const int lane = threadIdx.x & 31;
  const int h    = wave;
  const int jl   = lane & 15;
  const int j    = j0 + jl;
  const int sodd = lane >> 4;              // this lane covers s parity
  const float adj = a_d[j * H + h];
  const float weh = we[h];
  const float mej = mean_e[j];

  // ---- pass 1: online (max, sumexp) over sources for column j ----
  edge_tile_issue(edge, j0, 0, &evt[0][0][0], wave, lane);
  float m = -INFINITY, l = 0.f;
  for (int i = 0; i < NCH; ++i) {
    int s0 = i * 32;
    if (i + 1 < NCH) {
      edge_tile_issue(edge, j0, s0 + 32, &evt[(i + 1) & 1][0][0], wave, lane);
      edge_tile_wait_le1();                // tile i complete (in-order TDM)
    } else {
      edge_tile_wait_le0();
    }
    __syncthreads();                       // tile i visible to all waves
    const float* tile = &evt[i & 1][0][0];
#pragma unroll
    for (int t = 0; t < 16; ++t) {
      int sl = 2 * t + sodd;
      int s  = s0 + sl;
      float ev = tile[sl * 16 + jl];
      bool present; float evv;
      if (s == j) { present = true; evv = mej; }   // self-loop, fill='mean'
      else        { present = ev > 0.f; evv = ev; }
      if (present) {
        float x = a_s[s * H + h] + adj + evv * weh;
        x = x > 0.f ? x : 0.2f * x;                // leaky_relu
        float nm = fmaxf(m, x);
        l = l * __expf(m - nm) + __expf(x - nm);
        m = nm;
      }
    }
    __syncthreads();                       // tile i consumed by all waves
  }
  { // merge the two parity halves (lane ^ 16 has same j)
    float m2 = __shfl_xor(m, 16, 32);
    float l2 = __shfl_xor(l, 16, 32);
    float nm = fmaxf(m, m2);
    l = l * __expf(m - nm) + l2 * __expf(m2 - nm);
    m = nm;
  }
  const float inv_l = 1.0f / l;            // diagonal present -> l > 0

  // ---- pass 2: alpha tiles -> WMMA accumulation over all sources ----
  v8f acc[C / 16] = {};
  const int ah = sodd * 16;                // A K-offset (contiguous k = 16*sodd + s)
  const int bh = sodd * 16;                // B K-offset
  edge_tile_issue(edge, j0, 0, &evt[0][0][0], wave, lane);
  for (int i = 0; i < NCH; ++i) {
    int s0 = i * 32;
    if (i + 1 < NCH) {
      edge_tile_issue(edge, j0, s0 + 32, &evt[(i + 1) & 1][0][0], wave, lane);
      edge_tile_wait_le1();
    } else {
      edge_tile_wait_le0();
    }
    __syncthreads();                       // tile i ready
    const float* tile = &evt[i & 1][0][0];
#pragma unroll
    for (int t = 0; t < 16; ++t) {
      int sl = 2 * t + sodd;
      int s  = s0 + sl;
      float ev = tile[sl * 16 + jl];
      bool present; float evv;
      if (s == j) { present = true; evv = mej; }
      else        { present = ev > 0.f; evv = ev; }
      float alpha = 0.f;
      if (present) {
        float x = a_s[s * H + h] + adj + evv * weh;
        x = x > 0.f ? x : 0.2f * x;
        alpha = __expf(x - m) * inv_l;
      }
      alphaT[h][jl][sl] = (_Float16)alpha;
    }
    __syncthreads();                       // tile i consumed; alphaT complete
    v8h alo = *(const v8h*)&alphaT[h][jl][ah];
    v8h ahi = *(const v8h*)&alphaT[h][jl][8 + ah];
    v16h a = __builtin_shufflevector(alo, ahi, 0, 1, 2, 3, 4, 5, 6, 7,
                                     8, 9, 10, 11, 12, 13, 14, 15);
#pragma unroll
    for (int t = 0; t < C / 16; ++t) {
      v16h b = *(const v16h*)(hmsgT + (h * C + t * 16 + jl) * N_NODES + s0 + bh);
      acc[t] = __builtin_amdgcn_wmma_f32_16x16x32_f16(false, a, false, b, (short)0, acc[t], false, false);
    }
  }

  // ---- epilogue: bias + ELU ----
#pragma unroll
  for (int t = 0; t < C / 16; ++t) {
#pragma unroll
    for (int r = 0; r < 8; ++r) {
      int mm = j0 + r + (sodd ? 8 : 0);
      int nn = h * C + t * 16 + jl;
      float v = acc[t][r] + bias[nn];
      v = v > 0.f ? v : (__expf(v) - 1.0f);        // elu
      if (OUTF32) outf32[mm * HC + nn] = v;
      if (OUTF16) outf16[mm * HC + nn] = (_Float16)v;
    }
  }
}

// ---------------- deterministic pooling + final linear ---------------------
// One block per graph; fixed ascending node order -> bit-deterministic.
__global__ void k_pool(const float* __restrict__ h2, const int* __restrict__ bidx,
                       float* __restrict__ gsum, float* __restrict__ gcnt) {
  int b = blockIdx.x;          // graph
  int c = threadIdx.x;         // channel 0..63
  float s = 0.f; int cnt = 0;
  for (int n = 0; n < N_NODES; ++n) {
    if (bidx[n] == b) { s += h2[n * 64 + c]; ++cnt; }
  }
  gsum[b * 64 + c] = s;
  if (c == 0) gcnt[b] = (float)cnt;
}
__global__ void k_final(const float* __restrict__ gsum, const float* __restrict__ cnt,
                        const float* __restrict__ Wl, const float* __restrict__ bl,
                        float* __restrict__ out) {
  int t = threadIdx.x;
  if (t >= NUM_G * 10) return;
  int b = t / 10, o = t - b * 10;
  float cc = fmaxf(cnt[b], 1.0f);
  float s = bl[o];
  for (int c = 0; c < 64; ++c) s += (gsum[b * 64 + c] / cc) * Wl[c * 10 + o];
  out[b * 10 + o] = s;
}

// ---------------------------------------------------------------------------
extern "C" void kernel_launch(void* const* d_in, const int* in_sizes, int n_in,
                              void* d_out, int out_size, void* d_ws, size_t ws_size,
                              hipStream_t stream) {
  const float* x        = (const float*)d_in[0];
  const float* attn     = (const float*)d_in[1];
  const int*   bidx     = (const int*)  d_in[2];
  const float* w_agg    = (const float*)d_in[3];
  const float* b_agg    = (const float*)d_in[4];
  const float* W1       = (const float*)d_in[5];
  const float* att_src1 = (const float*)d_in[6];
  const float* att_dst1 = (const float*)d_in[7];
  const float* We1      = (const float*)d_in[8];
  const float* att_e1   = (const float*)d_in[9];
  const float* bias1    = (const float*)d_in[10];
  const float* W2       = (const float*)d_in[11];
  const float* att_src2 = (const float*)d_in[12];
  const float* att_dst2 = (const float*)d_in[13];
  const float* We2      = (const float*)d_in[14];
  const float* att_e2   = (const float*)d_in[15];
  const float* bias2    = (const float*)d_in[16];
  const float* W_lin    = (const float*)d_in[17];
  const float* b_lin    = (const float*)d_in[18];

  float* out_final = (float*)d_out;            // [8,10]
  float* edge      = (float*)d_out + 80;       // [2048,2048] edge_attr == masked agg

  // carve workspace (256B aligned)
  char* w = (char*)d_ws;
  auto alloc = [&](size_t bytes) -> void* {
    void* p = (void*)w;
    w += (bytes + 255) & ~(size_t)255;
    return p;
  };
  const int N = N_NODES, IN = 128, H1 = 4, C1 = 64, HC1 = 256, C2 = 64;
  _Float16* x_f16    = (_Float16*)alloc((size_t)N * IN * 2);
  _Float16* W1T_f16  = (_Float16*)alloc((size_t)HC1 * IN * 2);   // [256][128]
  _Float16* W2T_f16  = (_Float16*)alloc((size_t)C2 * HC1 * 2);   // [64][256]
  float*    h1_f32   = (float*)   alloc((size_t)N * HC1 * 4);
  _Float16* h1T_f16  = (_Float16*)alloc((size_t)HC1 * N * 2);    // messages^T layer 1
  float*    as1      = (float*)   alloc((size_t)N * H1 * 4);
  float*    ad1      = (float*)   alloc((size_t)N * H1 * 4);
  float*    we1      = (float*)   alloc(64);
  float*    we2      = (float*)   alloc(64);
  float*    mean_e   = (float*)   alloc((size_t)N * 4);
  _Float16* h1e_f16  = (_Float16*)alloc((size_t)N * HC1 * 2);    // elu(gat1), A of linear2
  float*    hh2_f32  = (float*)   alloc((size_t)N * C2 * 4);
  _Float16* hh2T_f16 = (_Float16*)alloc((size_t)C2 * N * 2);     // messages^T layer 2
  float*    as2      = (float*)   alloc((size_t)N * 4);
  float*    ad2      = (float*)   alloc((size_t)N * 4);
  float*    h2_f32   = (float*)   alloc((size_t)N * C2 * 4);
  float*    gsum     = (float*)   alloc((size_t)(NUM_G * 64 + NUM_G) * 4);
  float*    gcnt     = gsum + NUM_G * 64;

  // 1) agg + threshold -> edge_attr (also serves as mask & edge values)
  k_agg<<<(N * N) / 256, 256, 0, stream>>>(attn, w_agg, b_agg, edge);
  // 2) per-dst mean fill value for self-loops
  k_colstats<<<N / 256, 256, 0, stream>>>(edge, mean_e);
  // 3) f16 conversions (weights transposed) + per-head edge weights
  k_f32tof16<<<(N * IN + 255) / 256, 256, 0, stream>>>(x, x_f16, N * IN);
  k_f32tof16_T<<<(IN * HC1 + 255) / 256, 256, 0, stream>>>(W1, W1T_f16, IN, HC1);
  k_f32tof16_T<<<(HC1 * C2 + 255) / 256, 256, 0, stream>>>(W2, W2T_f16, HC1, C2);
  k_we<<<1, 32, 0, stream>>>(We1, att_e1, we1, H1, C1);
  k_we<<<1, 32, 0, stream>>>(We2, att_e2, we2, 1, C2);
  // 4) h1 = x @ W1   (WMMA; f32 for attdot, f16^T messages)
  k_gemm_f16<true, true><<<dim3(N / 16, HC1 / 16), 32, 0, stream>>>(
      x_f16, W1T_f16, h1_f32, h1T_f16, N, IN, HC1);
  // 5) attention scalars layer 1
  k_attdot<<<(N * H1) / 256, 256, 0, stream>>>(h1_f32, att_src1, att_dst1, as1, ad1, H1, C1);
  // 6) GAT layer 1: TDM-staged softmax + WMMA aggregation + bias + ELU
  k_gat_attn<4, 64, false, true><<<N / 16, 4 * 32, 0, stream>>>(
      edge, mean_e, as1, ad1, we1, h1T_f16, bias1, nullptr, h1e_f16);
  // 7) hh2 = h1e @ W2   (WMMA)
  k_gemm_f16<true, true><<<dim3(N / 16, C2 / 16), 32, 0, stream>>>(
      h1e_f16, W2T_f16, hh2_f32, hh2T_f16, N, HC1, C2);
  // 8) attention scalars layer 2
  k_attdot<<<N / 256, 256, 0, stream>>>(hh2_f32, att_src2, att_dst2, as2, ad2, 1, C2);
  // 9) GAT layer 2 -> f32 (+ ELU)
  k_gat_attn<1, 64, true, false><<<N / 16, 32, 0, stream>>>(
      edge, mean_e, as2, ad2, we2, hh2T_f16, bias2, h2_f32, nullptr);
  // 10) deterministic global mean pool + final linear
  k_pool<<<NUM_G, 64, 0, stream>>>(h2_f32, bidx, gsum, gcnt);
  k_final<<<1, 128, 0, stream>>>(gsum, gcnt, W_lin, b_lin, out_final);

  (void)in_sizes; (void)n_in; (void)out_size; (void)ws_size;
}
